// TAttention_56135222558855
// MI455X (gfx1250) — matrix operand order repaired and checked by reading, not compiled, not run on hardware
//
#include <hip/hip_runtime.h>
#include <hip/hip_bf16.h>

// ---------------------------------------------------------------------------
// CDNA5 / gfx1250 transformer encoder layer, bf16 WMMA everywhere.
// wave32; WMMA f32 16x16x32 bf16; double-buffered LDS staging via
// GLOBAL_LOAD_ASYNC_TO_LDS_B128 (ASYNCcnt) with one barrier per iteration;
// DPP16 row reductions for softmax.
// ---------------------------------------------------------------------------

typedef __attribute__((ext_vector_type(16))) __bf16 v16bf;
typedef __attribute__((ext_vector_type(8)))  float  v8f;

union U16frag {
    uint4  q[2];   // 32 bytes
    v16bf  v;
};

__device__ __forceinline__ unsigned short f2bf(float f) {
    union { float f; unsigned u; } v; v.f = f;
    unsigned r = v.u + 0x7FFFu + ((v.u >> 16) & 1u);   // round-to-nearest-even
    return (unsigned short)(r >> 16);
}

__device__ __forceinline__ v8f v8zero() {
    v8f z = {0.f,0.f,0.f,0.f,0.f,0.f,0.f,0.f};
    return z;
}

// ----- async global -> LDS copy engine (ASYNCcnt) ---------------------------
// LDS byte offset = low 32 bits of the generic (aperture-form) shared pointer.
__device__ __forceinline__ void async_g2l_b128(void* lds, const void* g) {
    asm volatile("global_load_async_to_lds_b128 %0, %1, off"
                 :: "v"((unsigned)(size_t)lds),
                    "v"((unsigned long long)(size_t)g)
                 : "memory");
}
__device__ __forceinline__ void wait_async0() {
    asm volatile("s_wait_asynccnt 0x0" ::: "memory");
}

// ----- DPP16 xor-reductions within 16-lane rows (masks 1,2,4,8) -------------
template<int CTRL>
__device__ __forceinline__ float dppf(float x) {
    return __int_as_float(
        __builtin_amdgcn_update_dpp(0, __float_as_int(x), CTRL, 0xf, 0xf, true));
}
__device__ __forceinline__ float redmax16(float x) {
    x = fmaxf(x, dppf<0xB1>(x));    // quad_perm(1,0,3,2)  : xor 1
    x = fmaxf(x, dppf<0x4E>(x));    // quad_perm(2,3,0,1)  : xor 2
    x = fmaxf(x, dppf<0x164>(x));   // row_xmask:4         : xor 4
    x = fmaxf(x, dppf<0x168>(x));   // row_xmask:8         : xor 8
    return x;
}
__device__ __forceinline__ float redsum16(float x) {
    x += dppf<0xB1>(x);
    x += dppf<0x4E>(x);
    x += dppf<0x164>(x);
    x += dppf<0x168>(x);
    return x;
}

// ---------------------------------------------------------------------------
// Weight transpose + cast: W (KxN f32, row major) -> WT (NxK bf16, row major)
// ---------------------------------------------------------------------------
__global__ __launch_bounds__(256) void transpose_cast_bf16(
    const float* __restrict__ W, unsigned short* __restrict__ WT, int K, int N)
{
    __shared__ unsigned short tile[32][33];
    int tx = threadIdx.x, ty = threadIdx.y;           // blockDim (32,8)
    int n = blockIdx.x * 32 + tx;
#pragma unroll
    for (int i = 0; i < 32; i += 8) {
        int k = blockIdx.y * 32 + ty + i;
        tile[ty + i][tx] = f2bf(W[(size_t)k * N + n]);
    }
    __syncthreads();
    int k2 = blockIdx.y * 32 + tx;
#pragma unroll
    for (int i = 0; i < 32; i += 8) {
        int n2 = blockIdx.x * 32 + ty + i;
        WT[(size_t)n2 * K + k2] = tile[tx][ty + i];
    }
}

// ---------------------------------------------------------------------------
// Row LayerNorm (D=1024) + cast to bf16.  One 256-thread block per row.
// ---------------------------------------------------------------------------
__global__ __launch_bounds__(256) void ln_cast_bf16(
    const float* __restrict__ X, const float* __restrict__ g,
    const float* __restrict__ b, unsigned short* __restrict__ Y, int D)
{
    int row = blockIdx.x;
    int t = threadIdx.x;
    const float4* xr = (const float4*)(X + (size_t)row * D);
    float4 v = xr[t];
    float s  = v.x + v.y + v.z + v.w;
    float s2 = v.x*v.x + v.y*v.y + v.z*v.z + v.w*v.w;
    // full-wave reduce: DPP within 16 lanes, then one cross-half shuffle
    s  = redsum16(s);
    s2 = redsum16(s2);
    s  += __shfl_xor(s,  16, 32);
    s2 += __shfl_xor(s2, 16, 32);
    __shared__ float rs[8], rs2[8];
    int w = t >> 5;
    if ((t & 31) == 0) { rs[w] = s; rs2[w] = s2; }
    __syncthreads();
    s = 0.f; s2 = 0.f;
#pragma unroll
    for (int i = 0; i < 8; i++) { s += rs[i]; s2 += rs2[i]; }
    float mean = s / (float)D;
    float var  = s2 / (float)D - mean * mean;
    float inv  = rsqrtf(var + 1e-5f);
    float4 gv = ((const float4*)g)[t];
    float4 bv = ((const float4*)b)[t];
    ushort4 o;
    o.x = f2bf((v.x - mean) * inv * gv.x + bv.x);
    o.y = f2bf((v.y - mean) * inv * gv.y + bv.y);
    o.z = f2bf((v.z - mean) * inv * gv.z + bv.z);
    o.w = f2bf((v.w - mean) * inv * gv.w + bv.w);
    ((ushort4*)(Y + (size_t)row * D))[t] = o;
}

// ---------------------------------------------------------------------------
// bf16 WMMA GEMM:  C[M,N] = A[M,K] * B[K,N]  with B supplied transposed (N x K).
// 256 threads / block, 8 waves as 4x2; block tile 128x128, wave tile 32x64.
// Async-copy double-buffered LDS (one barrier per K-chunk); stride 40 hw.
// ---------------------------------------------------------------------------
template<bool BIAS, bool RES, bool RELU, bool OUTF, bool OUTB, bool OUTBT>
__global__ __launch_bounds__(256, 2) void gemm_bf16_wmma(
    const unsigned short* __restrict__ A, const unsigned short* __restrict__ BT,
    const float* __restrict__ bias, const float* __restrict__ res,
    float* __restrict__ Cf, unsigned short* __restrict__ Cb,
    int M, int N, int K, float oscale)
{
    __shared__ unsigned short a_s[2][128 * 40];
    __shared__ unsigned short b_s[2][128 * 40];
    int t = threadIdx.x;
    int m0 = blockIdx.y * 128, n0 = blockIdx.x * 128;
    int lane = t & 31, wv = t >> 5, half = lane >> 4, l15 = lane & 15;
    int wr = wv >> 1, wc = wv & 1;

    auto stage = [&](int k0, int bufi) {
#pragma unroll
        for (int i = t; i < 512; i += 256) {         // 128 rows x 4 groups of 8
            int r = i >> 2, c = (i & 3) * 8;
            async_g2l_b128(&a_s[bufi][r * 40 + c], &A[(size_t)(m0 + r) * K + k0 + c]);
            async_g2l_b128(&b_s[bufi][r * 40 + c], &BT[(size_t)(n0 + r) * K + k0 + c]);
        }
    };

    v8f acc[2][4];
#pragma unroll
    for (int i = 0; i < 2; i++)
#pragma unroll
        for (int j = 0; j < 4; j++) acc[i][j] = v8zero();

    stage(0, 0);
    wait_async0();
    __syncthreads();

    int nk = K >> 5;
    for (int kk = 0; kk < nk; kk++) {
        if (kk + 1 < nk) stage((kk + 1) << 5, (kk + 1) & 1);

        const unsigned short* as = a_s[kk & 1];
        const unsigned short* bs = b_s[kk & 1];
        U16frag af[2], bf[4];
#pragma unroll
        for (int rg = 0; rg < 2; rg++) {
            const unsigned short* p = &as[(wr * 32 + rg * 16 + l15) * 40 + half * 8];
            af[rg].q[0] = *(const uint4*)p;
            af[rg].q[1] = *(const uint4*)(p + 16);
        }
#pragma unroll
        for (int cg = 0; cg < 4; cg++) {
            const unsigned short* p = &bs[(wc * 64 + cg * 16 + l15) * 40 + half * 8];
            bf[cg].q[0] = *(const uint4*)p;
            bf[cg].q[1] = *(const uint4*)(p + 16);
        }
#pragma unroll
        for (int rg = 0; rg < 2; rg++)
#pragma unroll
            for (int cg = 0; cg < 4; cg++)
                acc[rg][cg] = __builtin_amdgcn_wmma_f32_16x16x32_bf16(
                    false, af[rg].v, false, bf[cg].v, (short)0, acc[rg][cg], false, false);

        wait_async0();          // next tile's async copies landed in LDS
        __syncthreads();        // publish to all waves
    }

#pragma unroll
    for (int rg = 0; rg < 2; rg++)
#pragma unroll
        for (int cg = 0; cg < 4; cg++) {
            int gn = n0 + wc * 64 + cg * 16 + l15;
            float bsv = BIAS ? bias[gn] : 0.f;
#pragma unroll
            for (int r = 0; r < 8; r++) {
                int gm = m0 + wr * 32 + rg * 16 + r + 8 * half;
                float v = acc[rg][cg][r] * oscale + bsv;
                if (RES)  v += res[(size_t)gm * N + gn];
                if (RELU) v = fmaxf(v, 0.f);
                if (OUTF)  Cf[(size_t)gm * N + gn] = v;
                if (OUTB)  Cb[(size_t)gm * N + gn] = f2bf(v);
                if (OUTBT) Cb[(size_t)gn * M + gm] = f2bf(v);
            }
        }
}

// ---------------------------------------------------------------------------
// Flash attention, bf16 WMMA, Dh=64.  Grid: (S/128, B*H); 256 threads (8 waves,
// 16 q-rows each).  Online softmax with DPP reductions; P converted
// C-layout -> A-layout via wave-private LDS rows.  K/V^T double-buffered with
// async global->LDS copies.
// Q,K: row-major bf16 [B*S, D] head slices (Q pre-scaled by 1/sqrt(Dh));
// V: transposed bf16 [D, B*S].
// LDS pool (halfwords): [0,9216) Q-staging then P;
//   [9216 + buf*4608)   K buffers (buf = 0,1)
//   [18432 + buf*4608)  V buffers (buf = 0,1)
// ---------------------------------------------------------------------------
__global__ __launch_bounds__(256) void flash_attn_bf16(
    const unsigned short* __restrict__ Q, const unsigned short* __restrict__ Kb,
    const unsigned short* __restrict__ VT, unsigned short* __restrict__ O,
    int S, int Dm, int MT)
{
    const int HD = 64;
    __shared__ unsigned short smem[27648];

    int t = threadIdx.x;
    int lane = t & 31, wv = t >> 5, half = lane >> 4, l15 = lane & 15;
    int bh = blockIdx.y; int b = bh >> 4; int h = bh & 15;
    int qb = blockIdx.x * 128;

    size_t qbase = ((size_t)(b * S + qb)) * Dm + (size_t)h * HD;
    size_t kbase = ((size_t)(b * S)) * Dm + (size_t)h * HD;
    size_t vbase = ((size_t)(h * HD)) * MT + (size_t)b * S;

    auto stageKV = [&](int kt, int bufi) {
        unsigned short* kd = smem +  9216 + bufi * 4608;
        unsigned short* vd = smem + 18432 + bufi * 4608;
#pragma unroll
        for (int i = t; i < 512; i += 256) {         // 64 x 64 tiles
            int r = i >> 3, c = (i & 7) * 8;
            async_g2l_b128(&kd[r * 72 + c], &Kb[kbase + (size_t)(kt + r) * Dm + c]);
            async_g2l_b128(&vd[r * 72 + c], &VT[vbase + (size_t)r * MT + kt + c]);
        }
    };

    // stage Q tile (128 x 64) into smem[0,9216) and first K/V tile, one barrier
    for (int i = t; i < 1024; i += 256) {
        int r = i >> 3, c = (i & 7) * 8;
        async_g2l_b128(&smem[r * 72 + c], &Q[qbase + (size_t)r * Dm + c]);
    }
    stageKV(0, 0);
    wait_async0();
    __syncthreads();

    U16frag qa[2];
#pragma unroll
    for (int ch = 0; ch < 2; ch++) {
        const unsigned short* p = &smem[(wv * 16 + l15) * 72 + ch * 32 + half * 8];
        qa[ch].q[0] = *(const uint4*)p;
        qa[ch].q[1] = *(const uint4*)(p + 16);
    }

    float mrun[8], lrun[8];
    v8f oacc[4];
#pragma unroll
    for (int r = 0; r < 8; r++) { mrun[r] = -1e30f; lrun[r] = 0.f; }
#pragma unroll
    for (int cg = 0; cg < 4; cg++) oacc[cg] = v8zero();

    int nt = S / 64;
    for (int it = 0; it < nt; it++) {
        if (it + 1 < nt) stageKV((it + 1) * 64, (it + 1) & 1);
        const unsigned short* ks = smem +  9216 + (it & 1) * 4608;
        const unsigned short* vs = smem + 18432 + (it & 1) * 4608;

        // S = Q * K^T   (16 x 64 per wave; Q already carries 1/sqrt(Dh))
        v8f sa[4];
#pragma unroll
        for (int cg = 0; cg < 4; cg++) sa[cg] = v8zero();
#pragma unroll
        for (int ch = 0; ch < 2; ch++) {
#pragma unroll
            for (int cg = 0; cg < 4; cg++) {
                U16frag bfr;
                const unsigned short* p = &ks[(cg * 16 + l15) * 72 + ch * 32 + half * 8];
                bfr.q[0] = *(const uint4*)p;
                bfr.q[1] = *(const uint4*)(p + 16);
                sa[cg] = __builtin_amdgcn_wmma_f32_16x16x32_bf16(
                    false, qa[ch].v, false, bfr.v, (short)0, sa[cg], false, false);
            }
        }

        // online softmax: row r+8*half lives in one 16-lane half -> DPP reduce
#pragma unroll
        for (int r = 0; r < 8; r++) {
            float mt = fmaxf(fmaxf(sa[0][r], sa[1][r]), fmaxf(sa[2][r], sa[3][r]));
            mt = redmax16(mt);
            float mnew = fmaxf(mrun[r], mt);
            float alpha = __expf(mrun[r] - mnew);
            float rsum = 0.f;
#pragma unroll
            for (int cg = 0; cg < 4; cg++) {
                float pv = __expf(sa[cg][r] - mnew);
                sa[cg][r] = pv;
                rsum += pv;
            }
            rsum = redsum16(rsum);
            lrun[r] = lrun[r] * alpha + rsum;
            mrun[r] = mnew;
#pragma unroll
            for (int cg = 0; cg < 4; cg++) oacc[cg][r] *= alpha;
        }

        // write P (C layout -> LDS row-major), wave-private rows [wv*16, wv*16+16)
        // (aliases the Q staging area, whose data already lives in qa registers)
#pragma unroll
        for (int cg = 0; cg < 4; cg++)
#pragma unroll
            for (int r = 0; r < 8; r++)
                smem[(wv * 16 + r + 8 * half) * 72 + cg * 16 + l15] = f2bf(sa[cg][r]);
        asm volatile("s_wait_dscnt 0x0" ::: "memory");

        // O += P * V   (K = 64 keys)
#pragma unroll
        for (int ch = 0; ch < 2; ch++) {
            U16frag pa;
            const unsigned short* pp = &smem[(wv * 16 + l15) * 72 + ch * 32 + half * 8];
            pa.q[0] = *(const uint4*)pp;
            pa.q[1] = *(const uint4*)(pp + 16);
#pragma unroll
            for (int cg = 0; cg < 4; cg++) {
                U16frag vb;
                const unsigned short* vp = &vs[(cg * 16 + l15) * 72 + ch * 32 + half * 8];
                vb.q[0] = *(const uint4*)vp;
                vb.q[1] = *(const uint4*)(vp + 16);
                oacc[cg] = __builtin_amdgcn_wmma_f32_16x16x32_bf16(
                    false, pa.v, false, vb.v, (short)0, oacc[cg], false, false);
            }
        }
        wait_async0();          // next K/V tiles landed
        __syncthreads();
    }

    // normalize and emit O (bf16, row-major head slice)
#pragma unroll
    for (int cg = 0; cg < 4; cg++)
#pragma unroll
        for (int r = 0; r < 8; r++) {
            float val = oacc[cg][r] / lrun[r];
            size_t gm = (size_t)(b * S + qb + wv * 16 + r + 8 * half);
            O[gm * Dm + h * HD + cg * 16 + l15] = f2bf(val);
        }
}

// ---------------------------------------------------------------------------
// Host launcher
// ---------------------------------------------------------------------------
extern "C" void kernel_launch(void* const* d_in, const int* in_sizes, int n_in,
                              void* d_out, int out_size, void* d_ws, size_t ws_size,
                              hipStream_t stream) {
    const int D = 1024, Bb = 4, S = 2048, Hh = 16;
    const int MT = Bb * S;                       // 8192 rows total

    const float* x  = (const float*)d_in[0];
    const float* wq = (const float*)d_in[1];
    const float* wk = (const float*)d_in[2];
    const float* wv = (const float*)d_in[3];
    const float* wo = (const float*)d_in[4];
    const float* bo = (const float*)d_in[5];
    const float* gx = (const float*)d_in[6];
    const float* bx = (const float*)d_in[7];
    const float* gf = (const float*)d_in[8];
    const float* bf = (const float*)d_in[9];
    const float* w1 = (const float*)d_in[10];
    const float* b1 = (const float*)d_in[11];
    const float* w2 = (const float*)d_in[12];
    const float* b2 = (const float*)d_in[13];
    float* out = (float*)d_out;

    char* base = (char*)d_ws;
    size_t off = 0;
    auto take = [&](size_t bytes) -> char* {
        char* p = base + off;
        off += (bytes + 255) & ~(size_t)255;
        return p;
    };
    unsigned short* wqT = (unsigned short*)take((size_t)D * D * 2);
    unsigned short* wkT = (unsigned short*)take((size_t)D * D * 2);
    unsigned short* wvT = (unsigned short*)take((size_t)D * D * 2);
    unsigned short* woT = (unsigned short*)take((size_t)D * D * 2);
    unsigned short* w1T = (unsigned short*)take((size_t)D * D * 2);
    unsigned short* w2T = (unsigned short*)take((size_t)D * D * 2);
    unsigned short* xs  = (unsigned short*)take((size_t)MT * D * 2);
    unsigned short* qbf = (unsigned short*)take((size_t)MT * D * 2);
    unsigned short* kbf = (unsigned short*)take((size_t)MT * D * 2);
    unsigned short* vTt = (unsigned short*)take((size_t)MT * D * 2);
    unsigned short* obf = (unsigned short*)take((size_t)MT * D * 2);
    float*          xt  = (float*)        take((size_t)MT * D * 4);
    unsigned short* hbf = (unsigned short*)take((size_t)MT * D * 2);
    unsigned short* fbf = (unsigned short*)take((size_t)MT * D * 2);

    dim3 tb(32, 8), tg(D / 32, D / 32);
    transpose_cast_bf16<<<tg, tb, 0, stream>>>(wq, wqT, D, D);
    transpose_cast_bf16<<<tg, tb, 0, stream>>>(wk, wkT, D, D);
    transpose_cast_bf16<<<tg, tb, 0, stream>>>(wv, wvT, D, D);
    transpose_cast_bf16<<<tg, tb, 0, stream>>>(wo, woT, D, D);
    transpose_cast_bf16<<<tg, tb, 0, stream>>>(w1, w1T, D, D);
    transpose_cast_bf16<<<tg, tb, 0, stream>>>(w2, w2T, D, D);

    ln_cast_bf16<<<MT, 256, 0, stream>>>(x, gx, bx, xs, D);

    dim3 gg(D / 128, MT / 128);   // (8, 64)
    // Q (pre-scaled by 1/sqrt(Dh)), K row-major bf16; V stored transposed
    gemm_bf16_wmma<false,false,false,false,true ,false><<<gg, 256, 0, stream>>>(
        xs, wqT, nullptr, nullptr, nullptr, qbf, MT, D, D, 0.125f);
    gemm_bf16_wmma<false,false,false,false,true ,false><<<gg, 256, 0, stream>>>(
        xs, wkT, nullptr, nullptr, nullptr, kbf, MT, D, D, 1.0f);
    gemm_bf16_wmma<false,false,false,false,false,true ><<<gg, 256, 0, stream>>>(
        xs, wvT, nullptr, nullptr, nullptr, vTt, MT, D, D, 1.0f);

    flash_attn_bf16<<<dim3(S / 128, Bb * Hh), 256, 0, stream>>>(
        qbf, kbf, vTt, obf, S, D, MT);

    // xt = x + o @ wo + bo   (f32)
    gemm_bf16_wmma<true ,true ,false,true ,false,false><<<gg, 256, 0, stream>>>(
        obf, woT, bo, x, xt, nullptr, MT, D, D, 1.0f);

    ln_cast_bf16<<<MT, 256, 0, stream>>>(xt, gf, bf, hbf, D);

    // ffn1 = relu(h @ w1 + b1) -> bf16
    gemm_bf16_wmma<true ,false,true ,false,true ,false><<<gg, 256, 0, stream>>>(
        hbf, w1T, b1, nullptr, nullptr, fbf, MT, D, D, 1.0f);

    // out = xt + ffn1 @ w2 + b2
    gemm_bf16_wmma<true ,true ,false,true ,false,false><<<gg, 256, 0, stream>>>(
        fbf, w2T, b2, xt, out, nullptr, MT, D, D, 1.0f);
}
